// UMamba_83717502533810
// MI455X (gfx1250) — matrix operand order repaired and checked by reading, not compiled, not run on hardware
//
#include <hip/hip_runtime.h>
#include <hip/hip_bf16.h>
#include <math.h>

// ---------------------------------------------------------------------------
// MI455X (gfx1250) UMamba forward. All dense projections use
// v_wmma_f32_16x16x32_bf16 (wave32, f32 accumulate, 4 WMMA/wave/K-step with
// B-fragment reuse); selective scan is a per-(batch,channel) register
// recurrence; permutes / conv / LN are elementwise.
// Shape guarantees exploited by the GEMM: M % 64 == 0, lda % 4 == 0,
// weight-row stride K % 4 == 0 (true for every projection in this net);
// K % 32 == 0 everywhere except the dt-rank projection (templated edge path).
// ---------------------------------------------------------------------------

typedef __bf16 bf16;
typedef __attribute__((ext_vector_type(16))) __bf16 v16bf;
typedef __attribute__((ext_vector_type(8)))  float  v8f;

#define DIV_UP(a, b) (((a) + (b) - 1) / (b))

// round-to-nearest-even f32 -> bf16, two values packed into one u32
__device__ __forceinline__ unsigned pk2(float a, float b) {
  union { float f; unsigned u; } x, y;
  x.f = a; y.f = b;
  unsigned lo = (x.u + 0x7FFFu + ((x.u >> 16) & 1u)) >> 16;
  unsigned hi = (y.u + 0x7FFFu + ((y.u >> 16) & 1u)) & 0xFFFF0000u;
  return lo | hi;
}
__device__ __forceinline__ float siluf(float x) { return x / (1.f + __expf(-x)); }
__device__ __forceinline__ float geluf(float x) {
  float x3 = x * x * x;
  return 0.5f * x * (1.f + tanhf(0.7978845608028654f * (x + 0.044715f * x3)));
}
__device__ __forceinline__ float softplusf(float x) {
  return fmaxf(x, 0.f) + log1pf(__expf(-fabsf(x)));
}

union AFrag { uint4 q[2]; unsigned u[8]; v16bf v; };

// ---------------------------------------------------------------------------
// GEMM: C[M,N] = act(A[M,K] * W[N,K]^T + bias[N])
// 128 threads = 4 waves. Block tile 64(M) x 64(N); wave w owns 16 columns
// (n0 = blockIdx.y*64 + 16w) and all four 16-row slabs -> 4 WMMA per K-step,
// one B fragment reused across them. A tile staged in LDS as packed bf16
// pairs (u32): float4 global loads -> pk2 -> ds_store_b128; A fragments are
// two aligned uint4 (ds_load_b128) reads each.
// FK=true: K % 32 == 0 (branch-free fast path). FK=false: ragged-K edge.
// act: 0=none, 1=gelu(tanh), 2=softplus. Requires M%64==0, lda%4==0, K%4==0.
// ---------------------------------------------------------------------------
template <bool FK>
__global__ void k_gemm_wmma(const float* __restrict__ A, int lda,
                            const float* __restrict__ W,
                            const float* __restrict__ bias,
                            float* __restrict__ Cout, int ldc,
                            int M, int N, int K, int act) {
  __shared__ unsigned As[64][16];  // [m][k-pair], 4 KB
  const int lane = threadIdx.x & 31;
  const int wave = threadIdx.x >> 5;
  const int m0 = blockIdx.x * 64;
  const int n0 = blockIdx.y * 64 + wave * 16;
  const int ml = lane & 15;
  const int hi = lane >> 4;
  const int n = n0 + ml;
  const float* Wr = W + (long)n * K;

  v8f acc[4] = {{}, {}, {}, {}};
  for (int k0 = 0; k0 < K; k0 += 32) {
    __syncthreads();
    // stage A tile (64 rows x 32 k) as packed bf16 pairs; 2 tasks/thread
#pragma unroll
    for (int tsk = 0; tsk < 2; ++tsk) {
      int task = threadIdx.x + 128 * tsk;
      int row = task >> 2, q = task & 3;         // row 0..63, 8-k quad 0..3
      long base = (long)(m0 + row) * lda + k0 + 8 * q;
      uint4 pk;
      if (FK) {
        float4 f0 = *(const float4*)&A[base];
        float4 f1 = *(const float4*)&A[base + 4];
        pk.x = pk2(f0.x, f0.y); pk.y = pk2(f0.z, f0.w);
        pk.z = pk2(f1.x, f1.y); pk.w = pk2(f1.z, f1.w);
      } else {
        float f[8];
#pragma unroll
        for (int e = 0; e < 8; ++e) {
          int kk = k0 + 8 * q + e;
          f[e] = (kk < K) ? A[base + e] : 0.f;
        }
        pk.x = pk2(f[0], f[1]); pk.y = pk2(f[2], f[3]);
        pk.z = pk2(f[4], f[5]); pk.w = pk2(f[6], f[7]);
      }
      *(uint4*)&As[row][4 * q] = pk;             // ds_store_b128
    }
    __syncthreads();
    if (k0 + 32 < K)  // global_prefetch_b8 of next A K-slab
      __builtin_prefetch(&A[(long)(m0 + 2 * lane) * lda + k0 + 32], 0, 0);

    // B fragment (32x16): pair j holds K = 16*hi + 2j{,+1} of column n
    AFrag b;
    if (FK) {
      if (n < N) {
        const float* wp = Wr + k0 + 16 * hi;
#pragma unroll
        for (int j = 0; j < 4; ++j) {
          float4 f = *(const float4*)&wp[4 * j];
          b.u[2 * j + 0] = pk2(f.x, f.y);
          b.u[2 * j + 1] = pk2(f.z, f.w);
        }
      } else {
#pragma unroll
        for (int j = 0; j < 8; ++j) b.u[j] = 0u;
      }
    } else {
#pragma unroll
      for (int j = 0; j < 8; ++j) {
        int kk = k0 + 16 * hi + 2 * j;
        float b0 = (n < N && kk < K) ? Wr[kk] : 0.f;
        float b1 = (n < N && kk + 1 < K) ? Wr[kk + 1] : 0.f;
        b.u[j] = pk2(b0, b1);
      }
    }

    // 4 m-slabs; A frag = pairs {8hi..8hi+7} of the LDS row
#pragma unroll
    for (int mt = 0; mt < 4; ++mt) {
      AFrag a;
      a.q[0] = *(const uint4*)&As[16 * mt + ml][8 * hi];
      a.q[1] = *(const uint4*)&As[16 * mt + ml][8 * hi + 4];
      acc[mt] = __builtin_amdgcn_wmma_f32_16x16x32_bf16(
          false, a.v, false, b.v, (short)0, acc[mt], false, false);
    }
  }
  // D layout: VGPR r -> M = r + 8*hi, N = lane&15
  if (n < N) {
    float bv = bias ? bias[n] : 0.f;
#pragma unroll
    for (int mt = 0; mt < 4; ++mt)
#pragma unroll
      for (int r = 0; r < 8; ++r) {
        int m = m0 + 16 * mt + r + 8 * hi;
        float v = acc[mt][r] + bv;
        if (act == 1) v = geluf(v);
        else if (act == 2) v = softplusf(v);
        Cout[(long)m * ldc + n] = v;
      }
  }
}

// ---------------------------------------------------------------------------
// Zigzag path tables (fwd + inverse), p = flip_c*4 + flip_r*2 + transpose
// ---------------------------------------------------------------------------
__global__ void k_perms(int* __restrict__ fwd, int* __restrict__ rev, int N) {
  int total = 8 * N * N;
  int t = blockIdx.x * blockDim.x + threadIdx.x;
  if (t >= total) return;
  int p = t / (N * N);
  int k = t % (N * N);
  int transpose = p & 1, flip_r = (p >> 1) & 1, flip_c = (p >> 2) & 1;
  int r = k / N, c0 = k % N;
  int c = (r & 1) ? (N - 1 - c0) : c0;
  int aa = transpose ? c : r;
  int bb = transpose ? r : c;
  int i = flip_r ? (N - 1 - aa) : aa;
  int j = flip_c ? (N - 1 - bb) : bb;
  int v = i * N + j;
  fwd[p * N * N + k] = v;
  rev[p * N * N + v] = k;
}

// Patch embed (P=2, IN_CH=2) + sincos-2d positional embedding
__global__ void k_patch_pos(const float* __restrict__ x,
                            const float* __restrict__ pe_w,
                            const float* __restrict__ pe_b,
                            float* __restrict__ out) {
  int t = blockIdx.x * blockDim.x + threadIdx.x;
  if (t >= 4 * 4096 * 64) return;
  int o = t & 63;
  int m = (t >> 6) & 4095;
  int b = t >> 18;
  int gi = m >> 6, gj = m & 63;
  float s = 0.f;
#pragma unroll
  for (int c = 0; c < 2; ++c)
#pragma unroll
    for (int p = 0; p < 2; ++p)
#pragma unroll
      for (int q = 0; q < 2; ++q)
        s += x[((b * 2 + c) * 128 + (2 * gi + p)) * 128 + (2 * gj + q)] *
             pe_w[((o * 2 + c) * 2 + p) * 2 + q];
  s += pe_b[o];
  int kk = o & 15, sel = o >> 4;           // 0:sin(w) 1:cos(w) 2:sin(h) 3:cos(h)
  float omega = __powf(10000.f, -(float)kk / 16.f);
  float ang = ((sel < 2) ? (float)gj : (float)gi) * omega;
  s += (sel & 1) ? __cosf(ang) : __sinf(ang);
  out[t] = s;
}

// LayerNorm (g/b optional); one thread per row, C <= 1024
__global__ void k_ln(const float* __restrict__ x, float* __restrict__ y,
                     const float* __restrict__ g, const float* __restrict__ b,
                     int M, int C, float eps) {
  int r = blockIdx.x * blockDim.x + threadIdx.x;
  if (r >= M) return;
  const float* xr = x + (long)r * C;
  float m = 0.f;
  for (int c = 0; c < C; ++c) m += xr[c];
  m /= C;
  float v = 0.f;
  for (int c = 0; c < C; ++c) { float d = xr[c] - m; v += d * d; }
  v /= C;
  float inv = rsqrtf(v + eps);
  float* yr = y + (long)r * C;
  for (int c = 0; c < C; ++c) {
    float o = (xr[c] - m) * inv;
    if (g) o = o * g[c] + b[c];
    yr[c] = o;
  }
}

__global__ void k_gather(const float* __restrict__ x, float* __restrict__ t,
                         const int* __restrict__ perm, int B, int L, int C) {
  long i = (long)blockIdx.x * blockDim.x + threadIdx.x;
  if (i >= (long)B * L * C) return;
  int c = (int)(i % C);
  long rl = i / C;
  int l = (int)(rl % L), b = (int)(rl / L);
  t[i] = x[((long)b * L + perm[l]) * C + c];
}

__global__ void k_gather_add(float* __restrict__ x, const float* __restrict__ t,
                             const int* __restrict__ rev, int B, int L, int C) {
  long i = (long)blockIdx.x * blockDim.x + threadIdx.x;
  if (i >= (long)B * L * C) return;
  int c = (int)(i % C);
  long rl = i / C;
  int l = (int)(rl % L), b = (int)(rl / L);
  x[i] += t[((long)b * L + rev[l]) * C + c];
}

__global__ void k_add(float* __restrict__ x, const float* __restrict__ t, long n) {
  long i = (long)blockIdx.x * blockDim.x + threadIdx.x;
  if (i < n) x[i] += t[i];
}

// Causal depthwise conv (D_CONV=4) over x-half of xz, then SiLU
__global__ void k_conv_silu(const float* __restrict__ xz, int ld,
                            const float* __restrict__ cw,
                            const float* __restrict__ cb,
                            float* __restrict__ xc, int B, int L, int C) {
  long i = (long)blockIdx.x * blockDim.x + threadIdx.x;
  if (i >= (long)B * L * C) return;
  int d = (int)(i % C);
  long rl = i / C;
  int l = (int)(rl % L), b = (int)(rl / L);
  float s = cb[d];
#pragma unroll
  for (int j = 0; j < 4; ++j) {
    int ls = l - 3 + j;
    if (ls >= 0) s += xz[((long)b * L + ls) * ld + d] * cw[d * 4 + j];
  }
  xc[i] = siluf(s);
}

// Selective scan: one lane per (b, channel), 16 states in registers
__global__ void k_scan(const float* __restrict__ xc, const float* __restrict__ dt,
                       const float* __restrict__ dbc, int ldd, int dtr,
                       const float* __restrict__ xz, int ldz,
                       const float* __restrict__ A_log, const float* __restrict__ Dp,
                       float* __restrict__ y, int B, int L, int C) {
  int t = blockIdx.x * blockDim.x + threadIdx.x;
  if (t >= B * C) return;
  int d = t % C, b = t / C;
  float Ar[16], h[16];
#pragma unroll
  for (int n = 0; n < 16; ++n) { Ar[n] = -__expf(A_log[d * 16 + n]); h[n] = 0.f; }
  float Dv = Dp[d];
  for (int l = 0; l < L; ++l) {
    long row = (long)b * L + l;
    float dtv = dt[row * C + d];
    float xv = xc[row * C + d];
    const float* Bm = dbc + row * ldd + dtr;
    const float* Cm = Bm + 16;
    float dtx = dtv * xv;
    float acc = 0.f;
#pragma unroll
    for (int n = 0; n < 16; ++n) {
      float dA = __expf(dtv * Ar[n]);
      h[n] = h[n] * dA + dtx * Bm[n];
      acc += h[n] * Cm[n];
    }
    acc += Dv * xv;
    y[row * C + d] = acc * siluf(xz[row * ldz + C + d]);
  }
}

// PatchMerge rearrange (B,H,H,C)->(B,H/2*H/2,4C), order [00,10,01,11]
__global__ void k_merge(const float* __restrict__ x, float* __restrict__ out,
                        int B, int H, int C) {
  int Ho = H / 2;
  long i = (long)blockIdx.x * blockDim.x + threadIdx.x;
  if (i >= (long)B * Ho * Ho * 4 * C) return;
  int ch4 = (int)(i % (4 * C));
  long rest = i / (4 * C);
  int cc = (int)(rest % Ho); rest /= Ho;
  int r = (int)(rest % Ho);
  int b = (int)(rest / Ho);
  int q = ch4 / C, ch = ch4 % C;
  int dr = q & 1, dc = q >> 1;
  out[i] = x[(((long)b * H + (2 * r + dr)) * H + (2 * cc + dc)) * C + ch];
}

// PatchExpand rearrange: e(B,H,H,2C) -> out(B,2H,2H,C/2)
__global__ void k_expand_rearr(const float* __restrict__ e, float* __restrict__ out,
                               int B, int H, int C) {
  int c = C / 2, H2 = 2 * H;
  long i = (long)blockIdx.x * blockDim.x + threadIdx.x;
  if (i >= (long)B * H2 * H2 * c) return;
  int ch = (int)(i % c);
  long rest = i / c;
  int j2 = (int)(rest % H2); rest /= H2;
  int i2 = (int)(rest % H2);
  int b = (int)(rest / H2);
  int ii = i2 >> 1, p = i2 & 1, jj = j2 >> 1, q = j2 & 1;
  out[i] = e[(((long)b * H + ii) * H + jj) * (2 * C) + (p * 2 + q) * c + ch];
}

__global__ void k_concat(const float* __restrict__ a, const float* __restrict__ s,
                         float* __restrict__ out, long M, int c1, int c2) {
  long i = (long)blockIdx.x * blockDim.x + threadIdx.x;
  if (i >= M * (c1 + c2)) return;
  int j = (int)(i % (c1 + c2));
  long r = i / (c1 + c2);
  out[i] = (j < c1) ? a[r * c1 + j] : s[r * c2 + (j - c1)];
}

// fl-output (B,64,64,2,2,2) -> image, then 1x1 conv (2x2) head
__global__ void k_final(const float* __restrict__ f, const float* __restrict__ fc_w,
                        const float* __restrict__ fc_b, float* __restrict__ out) {
  int t = blockIdx.x * blockDim.x + threadIdx.x;
  if (t >= 4 * 2 * 128 * 128) return;
  int ww = t & 127, hh = (t >> 7) & 127, o = (t >> 14) & 1, b = t >> 15;
  int gi = hh >> 1, p = hh & 1, gj = ww >> 1, q = ww & 1;
  long base = (((long)b * 4096) + gi * 64 + gj) * 8 + (p * 2 + q) * 2;
  out[t] = fc_w[o * 2 + 0] * f[base + 0] + fc_w[o * 2 + 1] * f[base + 1] + fc_b[o];
}

// ---------------------------------------------------------------------------
// Host orchestration
// ---------------------------------------------------------------------------
struct BlockP {
  const float *in_w, *conv_w, *conv_b, *xproj_w, *dt_w, *dt_b, *A_log, *D,
      *out_w, *fc1_w, *fc1_b, *fc2_w, *fc2_b;
};
static BlockP grab_block(void* const* d_in, int& c) {
  BlockP p;
  p.in_w = (const float*)d_in[c++];   p.conv_w = (const float*)d_in[c++];
  p.conv_b = (const float*)d_in[c++]; p.xproj_w = (const float*)d_in[c++];
  p.dt_w = (const float*)d_in[c++];   p.dt_b = (const float*)d_in[c++];
  p.A_log = (const float*)d_in[c++];  p.D = (const float*)d_in[c++];
  p.out_w = (const float*)d_in[c++];  p.fc1_w = (const float*)d_in[c++];
  p.fc1_b = (const float*)d_in[c++];  p.fc2_w = (const float*)d_in[c++];
  p.fc2_b = (const float*)d_in[c++];
  return p;
}

static void gemm(hipStream_t s, const float* A, int lda, const float* W,
                 const float* bias, float* C, int ldc, int M, int N, int K, int act) {
  dim3 grid(M / 64, DIV_UP(N, 64));
  if ((K & 31) == 0)
    k_gemm_wmma<true><<<grid, 128, 0, s>>>(A, lda, W, bias, C, ldc, M, N, K, act);
  else
    k_gemm_wmma<false><<<grid, 128, 0, s>>>(A, lda, W, bias, C, ldc, M, N, K, act);
}

static void run_block(hipStream_t s, const BlockP& p, float* X, int B, int L,
                      int C, const int* fwd, const int* rev, float* T, float* XZ,
                      float* XC, float* DBC, float* DT, float* Y, float* MO,
                      float* H) {
  const int M = B * L, dtr = (C + 15) / 16, TPB = 256;
  long tot = (long)M * C;
  k_gather<<<DIV_UP(tot, TPB), TPB, 0, s>>>(X, T, fwd, B, L, C);
  k_ln<<<DIV_UP(M, 128), 128, 0, s>>>(T, T, nullptr, nullptr, M, C, 1e-6f);
  gemm(s, T, C, p.in_w, nullptr, XZ, 2 * C, M, 2 * C, C, 0);
  k_conv_silu<<<DIV_UP(tot, TPB), TPB, 0, s>>>(XZ, 2 * C, p.conv_w, p.conv_b, XC, B, L, C);
  gemm(s, XC, C, p.xproj_w, nullptr, DBC, dtr + 32, M, dtr + 32, C, 0);
  gemm(s, DBC, dtr + 32, p.dt_w, p.dt_b, DT, C, M, C, dtr, 2);   // softplus
  k_scan<<<DIV_UP(B * C, 128), 128, 0, s>>>(XC, DT, DBC, dtr + 32, dtr, XZ, 2 * C,
                                            p.A_log, p.D, Y, B, L, C);
  gemm(s, Y, C, p.out_w, nullptr, MO, C, M, C, C, 0);
  k_gather_add<<<DIV_UP(tot, TPB), TPB, 0, s>>>(X, MO, rev, B, L, C);
  k_ln<<<DIV_UP(M, 128), 128, 0, s>>>(X, T, nullptr, nullptr, M, C, 1e-6f);
  gemm(s, T, C, p.fc1_w, p.fc1_b, H, 4 * C, M, 4 * C, C, 1);     // gelu
  gemm(s, H, 4 * C, p.fc2_w, p.fc2_b, T, C, M, C, 4 * C, 0);
  k_add<<<DIV_UP(tot, TPB), TPB, 0, s>>>(X, T, tot);
}

extern "C" void kernel_launch(void* const* d_in, const int* in_sizes, int n_in,
                              void* d_out, int out_size, void* d_ws, size_t ws_size,
                              hipStream_t stream) {
  (void)in_sizes; (void)n_in; (void)out_size; (void)ws_size;
  const int B = 4;
  const float* x_in = (const float*)d_in[0];
  int cur = 1;
  auto nxt = [&]() { return (const float*)d_in[cur++]; };

  const float* pe_w = nxt();
  const float* pe_b = nxt();
  BlockP enc1[2]; for (int i = 0; i < 2; ++i) enc1[i] = grab_block(d_in, cur);
  const float *d1_ng = nxt(), *d1_nb = nxt(), *d1_red = nxt();
  BlockP d1b[2]; for (int i = 0; i < 2; ++i) d1b[i] = grab_block(d_in, cur);
  const float *d2_ng = nxt(), *d2_nb = nxt(), *d2_red = nxt();
  BlockP d2b[2]; for (int i = 0; i < 2; ++i) d2b[i] = grab_block(d_in, cur);
  const float *d3_ng = nxt(), *d3_nb = nxt(), *d3_red = nxt();
  BlockP d3b[2]; for (int i = 0; i < 2; ++i) d3b[i] = grab_block(d_in, cur);
  BlockP bot[2]; for (int i = 0; i < 2; ++i) bot[i] = grab_block(d_in, cur);
  const float *u3_exp = nxt(), *u3_ng = nxt(), *u3_nb = nxt();
  const float *u3_cvw = nxt(), *u3_cvb = nxt();
  BlockP u3b[2]; for (int i = 0; i < 2; ++i) u3b[i] = grab_block(d_in, cur);
  const float *u2_exp = nxt(), *u2_ng = nxt(), *u2_nb = nxt();
  const float *u2_cvw = nxt(), *u2_cvb = nxt();
  BlockP u2b[2]; for (int i = 0; i < 2; ++i) u2b[i] = grab_block(d_in, cur);
  const float *u1_exp = nxt(), *u1_ng = nxt(), *u1_nb = nxt();
  const float *u1_cvw = nxt(), *u1_cvb = nxt();
  BlockP u1b[2]; for (int i = 0; i < 2; ++i) u1b[i] = grab_block(d_in, cur);
  BlockP dec1[2]; for (int i = 0; i < 2; ++i) dec1[i] = grab_block(d_in, cur);
  const float *fl_w = nxt(), *fl_b = nxt(), *fc_w = nxt(), *fc_b = nxt();

  // ---- workspace layout: int perm tables, then float arena -------------
  int* pi = (int*)d_ws;
  int* fwd64 = pi;               int* rev64 = fwd64 + 8 * 4096;
  int* fwd32 = rev64 + 8 * 4096; int* rev32 = fwd32 + 8 * 1024;
  int* fwd16 = rev32 + 8 * 1024; int* rev16 = fwd16 + 8 * 256;
  int* fwd8  = rev16 + 8 * 256;  int* rev8  = fwd8 + 8 * 64;
  float* arena = (float*)(rev8 + 8 * 64);
  float* X = arena;              // 1,048,576
  float* T = X + 1048576;        // 1,048,576
  float* XZ = T + 1048576;       // 2,097,152
  float* XC = XZ + 2097152;      // 1,048,576
  float* DBC = XC + 1048576;     // 1,048,576
  float* DT = DBC + 1048576;     // 1,048,576
  float* Y = DT + 1048576;       // 1,048,576
  float* MO = Y + 1048576;       // 1,048,576
  float* H = MO + 1048576;       // 4,194,304 (also E / CAT scratch in up path)
  float* S1 = H + 4194304;       // 1,048,576
  float* S2 = S1 + 1048576;      // 524,288
  float* S3 = S2 + 524288;       // 262,144
  float* FB = S3 + 262144;       // 131,072
  float* E = H;                  // expand gemm out (<= 1M floats)
  float* CAT = H + 1048576;      // concat buf (<= 2M floats)

  // ---- constants: zigzag permutations + patch embed w/ pos embed -------
  k_perms<<<DIV_UP(8 * 4096, 256), 256, 0, stream>>>(fwd64, rev64, 64);
  k_perms<<<DIV_UP(8 * 1024, 256), 256, 0, stream>>>(fwd32, rev32, 32);
  k_perms<<<DIV_UP(8 * 256, 256), 256, 0, stream>>>(fwd16, rev16, 16);
  k_perms<<<DIV_UP(8 * 64, 256), 256, 0, stream>>>(fwd8, rev8, 8);
  k_patch_pos<<<DIV_UP(4 * 4096 * 64, 256), 256, 0, stream>>>(x_in, pe_w, pe_b, X);

  auto blk = [&](const BlockP& p, int L, int C, const int* fwd, const int* rev,
                 int path) {
    run_block(stream, p, X, B, L, C, fwd + path * L, rev + path * L, T, XZ, XC,
              DBC, DT, Y, MO, H);
  };
  auto merge = [&](const float* ng, const float* nb, const float* red, int Hs, int C) {
    int Ho = Hs / 2, M = B * Ho * Ho;
    long tot = (long)M * 4 * C;
    k_merge<<<DIV_UP(tot, 256), 256, 0, stream>>>(X, XZ, B, Hs, C);
    k_ln<<<DIV_UP(M, 128), 128, 0, stream>>>(XZ, XZ, ng, nb, M, 4 * C, 1e-5f);
    gemm(stream, XZ, 4 * C, red, nullptr, X, 2 * C, M, 2 * C, 4 * C, 0);
  };
  auto up = [&](const float* expw, const float* ng, const float* nb,
                const float* cvw, const float* cvb, const float* skip,
                int Hs, int C) {
    int L = Hs * Hs, M = B * L, c = C / 2;
    gemm(stream, X, C, expw, nullptr, E, 2 * C, M, 2 * C, C, 0);
    long tot = (long)B * 4 * L * c;
    k_expand_rearr<<<DIV_UP(tot, 256), 256, 0, stream>>>(E, XZ, B, Hs, C);
    k_ln<<<DIV_UP(B * 4 * L, 128), 128, 0, stream>>>(XZ, XZ, ng, nb, B * 4 * L, c, 1e-5f);
    long M2 = (long)B * 4 * L;
    k_concat<<<DIV_UP(M2 * 2 * c, 256), 256, 0, stream>>>(XZ, skip, CAT, M2, c, c);
    gemm(stream, CAT, 2 * c, cvw, cvb, X, c, (int)M2, c, 2 * c, 0);
  };

  // ---- encoder ----------------------------------------------------------
  blk(enc1[0], 4096, 64, fwd64, rev64, 0);
  blk(enc1[1], 4096, 64, fwd64, rev64, 1);
  hipMemcpyAsync(S1, X, (size_t)B * 4096 * 64 * 4, hipMemcpyDeviceToDevice, stream);
  merge(d1_ng, d1_nb, d1_red, 64, 64);
  blk(d1b[0], 1024, 128, fwd32, rev32, 0);
  blk(d1b[1], 1024, 128, fwd32, rev32, 1);
  hipMemcpyAsync(S2, X, (size_t)B * 1024 * 128 * 4, hipMemcpyDeviceToDevice, stream);
  merge(d2_ng, d2_nb, d2_red, 32, 128);
  blk(d2b[0], 256, 256, fwd16, rev16, 0);
  blk(d2b[1], 256, 256, fwd16, rev16, 1);
  hipMemcpyAsync(S3, X, (size_t)B * 256 * 256 * 4, hipMemcpyDeviceToDevice, stream);
  merge(d3_ng, d3_nb, d3_red, 16, 256);
  blk(d3b[0], 64, 512, fwd8, rev8, 0);
  blk(d3b[1], 64, 512, fwd8, rev8, 1);
  // ---- bottleneck -------------------------------------------------------
  blk(bot[0], 64, 512, fwd8, rev8, 0);
  blk(bot[1], 64, 512, fwd8, rev8, 1);
  // ---- decoder ----------------------------------------------------------
  up(u3_exp, u3_ng, u3_nb, u3_cvw, u3_cvb, S3, 8, 512);
  blk(u3b[0], 256, 256, fwd16, rev16, 0);
  blk(u3b[1], 256, 256, fwd16, rev16, 1);
  up(u2_exp, u2_ng, u2_nb, u2_cvw, u2_cvb, S2, 16, 256);
  blk(u2b[0], 1024, 128, fwd32, rev32, 0);
  blk(u2b[1], 1024, 128, fwd32, rev32, 1);
  up(u1_exp, u1_ng, u1_nb, u1_cvw, u1_cvb, S1, 32, 128);
  blk(u1b[0], 4096, 64, fwd64, rev64, 0);
  blk(u1b[1], 4096, 64, fwd64, rev64, 1);
  blk(dec1[0], 4096, 64, fwd64, rev64, 0);
  blk(dec1[1], 4096, 64, fwd64, rev64, 1);
  // ---- head -------------------------------------------------------------
  gemm(stream, X, 64, fl_w, fl_b, FB, 8, B * 4096, 8, 64, 0);
  k_final<<<DIV_UP(4 * 2 * 128 * 128, 256), 256, 0, stream>>>(FB, fc_w, fc_b,
                                                              (float*)d_out);
}
